// TGATUNet_49134425866402
// MI455X (gfx1250) — matrix operand order repaired and checked by reading, not compiled, not run on hardware
//
#include <hip/hip_runtime.h>
#include <math.h>

// ---------------------------------------------------------------------------
// MI455X (gfx1250) implementation of the TGAT-UNet reference.
// Dense math: v_wmma_f32_16x16x32_bf16 (f32 in HBM, bf16 LDS tiles, f32 acc),
// double-buffered LDS staging, packed bf16 conversion, async global->LDS copy
// for the top-k row cache. wave32, 256-thread blocks (8 waves).
// ---------------------------------------------------------------------------

#define NN 4096
#define ACT_NONE      0
#define ACT_RELU      1
#define ACT_TANH      2
#define ACT_RELU_DIAG 3   // relu, then diagonal := -1e9 (graph-learn mask)

#define GLB_AS __attribute__((address_space(1)))
#define LDS_AS __attribute__((address_space(3)))

typedef __attribute__((ext_vector_type(16))) __bf16 v16bf;
typedef __attribute__((ext_vector_type(8)))  float  v8f;
typedef __attribute__((ext_vector_type(4)))  int    v4i;

union FragBF16 {
  uint4 u[2];
  v16bf v;
};

__device__ __forceinline__ unsigned short f2bf(float f) {
  unsigned int u = __float_as_uint(f);
  unsigned int r = (u >> 16) & 1u;           // round-to-nearest-even
  u += 0x7FFFu + r;
  return (unsigned short)(u >> 16);
}

// Pack two f32 -> two bf16 in one dword (v_cvt_pk_bf16_f32 when available).
__device__ __forceinline__ unsigned int pk2bf(float a, float b) {
#if __has_builtin(__builtin_amdgcn_cvt_pk_bf16_f32)
  auto p = __builtin_amdgcn_cvt_pk_bf16_f32(a, b);
  unsigned int u;
  __builtin_memcpy(&u, &p, 4);
  return u;
#else
  return (unsigned int)f2bf(a) | ((unsigned int)f2bf(b) << 16);
#endif
}

// ---------------------------------------------------------------------------
// Tiled WMMA GEMM:  C = act(alpha*(A@B[T]) + bias + resid)
//   A: M x K  (row stride lda, f32)
//   B: transB==0 -> K x N (row stride ldb);  transB==1 -> N x K (row stride ldb)
//   C: M x N  (row stride ldc, f32); resid indexed with ldc as well.
// Block = 64x64 tile, BK=32, 256 threads = 8 waves, wave -> 16x32 of C.
// LDS tiles are double-buffered: one barrier per K-step, next tile's
// global_load_b128s overlap the current tile's WMMAs.
// ---------------------------------------------------------------------------
__global__ __launch_bounds__(256) void wmma_gemm_bf16(
    const float* __restrict__ A, int lda,
    const float* __restrict__ B, int ldb, int transB,
    float* __restrict__ C, int ldc,
    int M, int N, int K,
    const float* __restrict__ bias,
    const float* __restrict__ resid,
    float alpha, int act)
{
  __shared__ __align__(16) unsigned short As[2][64][32];
  __shared__ __align__(16) unsigned short Bs[2][64][32];   // Bs[n][k]

  const int tid  = threadIdx.x;
  const int lane = tid & 31;
  const int wv   = tid >> 5;           // 0..7
  const int row0 = blockIdx.y * 64;
  const int col0 = blockIdx.x * 64;
  const int tm   = wv & 3;             // wave tile-row  (16 rows each)
  const int tn2  = wv >> 2;            // wave tile-col pair (32 cols each)

  v8f acc0 = {};
  v8f acc1 = {};

  auto stageA = [&](int buf, int k0) {
    const int cg = tid & 7;            // 4-col group within BK=32
    const int r0 = tid >> 3;           // 0..31
    #pragma unroll
    for (int p = 0; p < 2; p++) {
      const int r = r0 + p * 32;
      const float4 f = *(const float4*)(A + (size_t)(row0 + r) * lda + k0 + cg * 4);
      *(uint2*)&As[buf][r][cg * 4] = make_uint2(pk2bf(f.x, f.y), pk2bf(f.z, f.w));
    }
  };
  auto stageB = [&](int buf, int k0) {
    if (transB) {
      const int cg = tid & 7;
      const int n0 = tid >> 3;
      #pragma unroll
      for (int p = 0; p < 2; p++) {
        const int n = n0 + p * 32;
        const float4 f = *(const float4*)(B + (size_t)(col0 + n) * ldb + k0 + cg * 4);
        *(uint2*)&Bs[buf][n][cg * 4] = make_uint2(pk2bf(f.x, f.y), pk2bf(f.z, f.w));
      }
    } else {
      const int ng  = tid & 15;        // 4-col group within 64 N cols
      const int kk0 = tid >> 4;        // 0..15
      #pragma unroll
      for (int p = 0; p < 2; p++) {
        const int kk = kk0 + p * 16;
        const float4 f = *(const float4*)(B + (size_t)(k0 + kk) * ldb + col0 + ng * 4);
        Bs[buf][ng * 4 + 0][kk] = f2bf(f.x);
        Bs[buf][ng * 4 + 1][kk] = f2bf(f.y);
        Bs[buf][ng * 4 + 2][kk] = f2bf(f.z);
        Bs[buf][ng * 4 + 3][kk] = f2bf(f.w);
      }
    }
  };

  stageA(0, 0);
  stageB(0, 0);
  const int nk = K >> 5;

  for (int kb = 0; kb < nk; kb++) {
    __syncthreads();
    if (kb + 1 < nk) {                 // prefetch next tile into other buffer
      stageA((kb + 1) & 1, (kb + 1) << 5);
      stageB((kb + 1) & 1, (kb + 1) << 5);
    }
    const int cbuf = kb & 1;

    // ---- build fragments, contiguous k = 16*(lane/16) + s in A and B (wave32) ----
    FragBF16 af, bf0, bf1;
    {
      const int arow = tm * 16 + (lane & 15);
      const int g    = lane >> 4;                 // K chunk group
      const uint4* ap = reinterpret_cast<const uint4*>(&As[cbuf][arow][0]);
      af.u[0] = ap[g * 2];        // K = g*16 .. g*16+7
      af.u[1] = ap[g * 2 + 1];    // K = g*16+8 .. g*16+15
    }
    {
      const int bcol = tn2 * 32 + (lane & 15);
      const int kh   = lane >> 4;                 // K half
      const uint4* bp0 = reinterpret_cast<const uint4*>(&Bs[cbuf][bcol][0]);
      bf0.u[0] = bp0[kh * 2];
      bf0.u[1] = bp0[kh * 2 + 1];
      const uint4* bp1 = reinterpret_cast<const uint4*>(&Bs[cbuf][bcol + 16][0]);
      bf1.u[0] = bp1[kh * 2];
      bf1.u[1] = bp1[kh * 2 + 1];
    }

    acc0 = __builtin_amdgcn_wmma_f32_16x16x32_bf16(false, af.v, false, bf0.v,
                                                   (short)0, acc0, false, false);
    acc1 = __builtin_amdgcn_wmma_f32_16x16x32_bf16(false, af.v, false, bf1.v,
                                                   (short)0, acc1, false, false);
  }

  // ---- epilogue (C/D layout: VGPR i -> M = i + 8*(lane/16), N = lane%16) ----
  const int rbase = row0 + tm * 16 + ((lane >> 4) << 3);
  const int cb    = col0 + tn2 * 32 + (lane & 15);
  #pragma unroll
  for (int i = 0; i < 8; i++) {
    const int r = rbase + i;
    #pragma unroll
    for (int half = 0; half < 2; half++) {
      const int c = cb + half * 16;
      float v = alpha * (half == 0 ? acc0[i] : acc1[i]);
      if (bias)  v += bias[c];
      if (resid) v += resid[(size_t)r * ldc + c];
      if (act == ACT_RELU)           v = v > 0.f ? v : 0.f;
      else if (act == ACT_TANH)      v = tanhf(v);
      else if (act == ACT_RELU_DIAG) { v = v > 0.f ? v : 0.f; if (r == c) v = -1e9f; }
      C[(size_t)r * ldc + c] = v;
    }
  }
}

// ---------------------------------------------------------------------------
// Per-row top-16 by iterative argmax. Row cached in LDS via CDNA5 async
// global->LDS copy (ASYNCcnt) when available; 16KB << 320KB/WGP.
// Matches jax.lax.top_k ordering (desc value, lowest index on ties).
// ---------------------------------------------------------------------------
__global__ __launch_bounds__(256) void topk16_kernel(const float* __restrict__ S,
                                                     int* __restrict__ idx)
{
  __shared__ __align__(16) float vals[NN];
  __shared__ float rv[256];
  __shared__ int   ri[256];
  const int row = blockIdx.x, t = threadIdx.x;

#if __has_builtin(__builtin_amdgcn_global_load_async_to_lds_b128)
  {
    const float* gsrc = S + (size_t)row * NN;
    #pragma unroll
    for (int i = 0; i < 4; i++) {
      const int e = (i * 256 + t) * 4;           // 16B chunk per lane
      __builtin_amdgcn_global_load_async_to_lds_b128(
          (GLB_AS v4i*)(void*)(gsrc + e),
          (LDS_AS v4i*)(void*)&vals[e], 0, 0);
    }
    asm volatile("s_wait_asynccnt 0x0" ::: "memory");
  }
  __syncthreads();
#else
  for (int i = t; i < NN; i += 256) vals[i] = S[(size_t)row * NN + i];
  __syncthreads();
#endif

  for (int sel = 0; sel < 16; sel++) {
    float bv = -3.4e38f; int bi = 0x7fffffff;
    for (int i = t; i < NN; i += 256) {
      const float v = vals[i];
      if (v > bv) { bv = v; bi = i; }
    }
    rv[t] = bv; ri[t] = bi;
    __syncthreads();
    for (int off = 128; off > 0; off >>= 1) {
      if (t < off) {
        if (rv[t + off] > rv[t] || (rv[t + off] == rv[t] && ri[t + off] < ri[t])) {
          rv[t] = rv[t + off]; ri[t] = ri[t + off];
        }
      }
      __syncthreads();
    }
    if (t == 0) { idx[row * 16 + sel] = ri[0]; vals[ri[0]] = -3.4e38f; }
    __syncthreads();
  }
}

// ---------------------------------------------------------------------------
// GAT: per-node, per-head attention logits  as = <h, a_src>, ad = <h, a_dst>
// ---------------------------------------------------------------------------
__global__ __launch_bounds__(256) void gat_prep_kernel(
    const float* __restrict__ h, const float* __restrict__ a_s,
    const float* __restrict__ a_d, float* __restrict__ as_o,
    float* __restrict__ ad_o, int H)
{
  const int n = blockIdx.x, c = threadIdx.x;
  const int F = 256 / H;
  __shared__ float ss[256], sd[256];
  const float x = h[(size_t)n * 256 + c];
  ss[c] = x * a_s[c];
  sd[c] = x * a_d[c];
  __syncthreads();
  for (int off = F >> 1; off > 0; off >>= 1) {
    if ((c & (F - 1)) < off) { ss[c] += ss[c + off]; sd[c] += sd[c + off]; }
    __syncthreads();
  }
  if ((c & (F - 1)) == 0) {
    const int hh = c / F;
    as_o[n * H + hh] = ss[c];
    ad_o[n * H + hh] = sd[c];
  }
}

// ---------------------------------------------------------------------------
// GAT aggregation: node n has exactly 17 in-edges (16 top-k sources + self).
// e = leaky_relu(as[src]+ad[n], 0.2); softmax over 17; out = sum alpha*h[src]+b
// Output strides (osn, osc) let the last layer write out.T directly.
// ---------------------------------------------------------------------------
__global__ __launch_bounds__(256) void gat_agg_kernel(
    const float* __restrict__ h, const int* __restrict__ idx,
    const float* __restrict__ as_, const float* __restrict__ ad_,
    const float* __restrict__ bias, float* __restrict__ out,
    int H, int doRelu, int osn, int osc)
{
  const int n = blockIdx.x, c = threadIdx.x;
  const int F = 256 / H;
  __shared__ int   srcs[17];
  __shared__ float alpha[4 * 17];
  if (c < 16)       srcs[c]  = idx[n * 16 + c];
  else if (c == 16) srcs[16] = n;
  __syncthreads();
  if (c < 17 * H) {
    const int j = c % 17, hh = c / 17;
    const float e = as_[srcs[j] * H + hh] + ad_[n * H + hh];
    alpha[hh * 17 + j] = e > 0.f ? e : 0.2f * e;
  }
  __syncthreads();
  if (c < H) {
    float m = -3.4e38f;
    for (int j = 0; j < 17; j++) m = fmaxf(m, alpha[c * 17 + j]);
    float s = 0.f;
    for (int j = 0; j < 17; j++) {
      const float p = __expf(alpha[c * 17 + j] - m);
      alpha[c * 17 + j] = p; s += p;
    }
    const float inv = 1.f / s;
    for (int j = 0; j < 17; j++) alpha[c * 17 + j] *= inv;
  }
  __syncthreads();
  const int hh = c / F;
  float acc = 0.f;
  #pragma unroll
  for (int j = 0; j < 17; j++)
    acc += alpha[hh * 17 + j] * h[(size_t)srcs[j] * 256 + c];
  acc += bias[c];
  if (doRelu) acc = fmaxf(acc, 0.f);
  out[(size_t)n * osn + (size_t)c * osc] = acc;
}

// ---------------------------------------------------------------------------
// Row softmax over 4096 columns (with pre-softmax scale), in place.
// ---------------------------------------------------------------------------
__global__ __launch_bounds__(256) void softmax_row_kernel(float* __restrict__ S,
                                                          int ncols, float scale)
{
  const int row = blockIdx.x, t = threadIdx.x;
  __shared__ float red[256];
  float v[16];
  float m = -3.4e38f;
  #pragma unroll
  for (int i = 0; i < 16; i++) {
    v[i] = S[(size_t)row * ncols + i * 256 + t] * scale;
    m = fmaxf(m, v[i]);
  }
  red[t] = m; __syncthreads();
  for (int off = 128; off > 0; off >>= 1) {
    if (t < off) red[t] = fmaxf(red[t], red[t + off]);
    __syncthreads();
  }
  m = red[0]; __syncthreads();
  float s = 0.f;
  #pragma unroll
  for (int i = 0; i < 16; i++) { v[i] = __expf(v[i] - m); s += v[i]; }
  red[t] = s; __syncthreads();
  for (int off = 128; off > 0; off >>= 1) {
    if (t < off) red[t] += red[t + off];
    __syncthreads();
  }
  const float inv = 1.f / red[0];
  #pragma unroll
  for (int i = 0; i < 16; i++)
    S[(size_t)row * ncols + i * 256 + t] = v[i] * inv;
}

// ---------------------------------------------------------------------------
// LayerNorm over 256 features, in place.
// ---------------------------------------------------------------------------
__global__ __launch_bounds__(256) void layernorm_kernel(float* __restrict__ X,
                                                        const float* __restrict__ g,
                                                        const float* __restrict__ b)
{
  const int row = blockIdx.x, t = threadIdx.x;
  __shared__ float red[256];
  const float v = X[(size_t)row * 256 + t];
  red[t] = v; __syncthreads();
  for (int off = 128; off > 0; off >>= 1) {
    if (t < off) red[t] += red[t + off];
    __syncthreads();
  }
  const float mu = red[0] * (1.f / 256.f);
  __syncthreads();
  const float d = v - mu;
  red[t] = d * d; __syncthreads();
  for (int off = 128; off > 0; off >>= 1) {
    if (t < off) red[t] += red[t + off];
    __syncthreads();
  }
  const float var = red[0] * (1.f / 256.f);
  X[(size_t)row * 256 + t] = d * rsqrtf(var + 1e-5f) * g[t] + b[t];
}

// ---------------------------------------------------------------------------
// Host sequencing
// ---------------------------------------------------------------------------
static void launch_gemm(const float* A, int lda, const float* B, int ldb, int transB,
                        float* C, int ldc, int M, int N, int K,
                        const float* bias, const float* resid,
                        float alpha, int act, hipStream_t s)
{
  dim3 grid(N / 64, M / 64);
  wmma_gemm_bf16<<<grid, 256, 0, s>>>(A, lda, B, ldb, transB, C, ldc,
                                      M, N, K, bias, resid, alpha, act);
}

extern "C" void kernel_launch(void* const* d_in, const int* in_sizes, int n_in,
                              void* d_out, int out_size, void* d_ws, size_t ws_size,
                              hipStream_t stream)
{
  (void)in_sizes; (void)n_in; (void)out_size; (void)ws_size;

  const float* window   = (const float*)d_in[0];
  const float* gl_w     = (const float*)d_in[1];
  const float* gl_b     = (const float*)d_in[2];
  const float* enc_W    = (const float*)d_in[3];
  const float* enc_asrc = (const float*)d_in[4];
  const float* enc_adst = (const float*)d_in[5];
  const float* enc_b    = (const float*)d_in[6];
  const float* tr_wqkv  = (const float*)d_in[7];
  const float* tr_bqkv  = (const float*)d_in[8];
  const float* tr_wo    = (const float*)d_in[9];
  const float* tr_bo    = (const float*)d_in[10];
  const float* tr_ln1_g = (const float*)d_in[11];
  const float* tr_ln1_b = (const float*)d_in[12];
  const float* tr_w1    = (const float*)d_in[13];
  const float* tr_b1    = (const float*)d_in[14];
  const float* tr_w2    = (const float*)d_in[15];
  const float* tr_b2    = (const float*)d_in[16];
  const float* tr_ln2_g = (const float*)d_in[17];
  const float* tr_ln2_b = (const float*)d_in[18];
  const float* skip_w   = (const float*)d_in[19];
  const float* skip_b   = (const float*)d_in[20];
  const float* dec_W    = (const float*)d_in[21];
  const float* dec_asrc = (const float*)d_in[22];
  const float* dec_adst = (const float*)d_in[23];
  const float* dec_b    = (const float*)d_in[24];
  const float* dec_lW   = (const float*)d_in[25];
  const float* dec_lasrc= (const float*)d_in[26];
  const float* dec_ladst= (const float*)d_in[27];
  const float* dec_lb   = (const float*)d_in[28];

  // ---- workspace carve-out (floats) ----
  float* W = (float*)d_ws;
  size_t o = 0;
  float* SIM  = W + o; o += (size_t)NN * NN;   // 64MB: similarity / per-head scores
  float* GLH  = W + o; o += (size_t)NN * 256;  // tanh graph-learn features
  float* X0   = W + o; o += (size_t)NN * 256;  // activation ping
  float* X1   = W + o; o += (size_t)NN * 256;  // activation pong
  float* HLIN = W + o; o += (size_t)NN * 768;  // linear scratch (qkv / gat h / ffn)
  float* OHB  = W + o; o += (size_t)NN * 256;  // merged attention output
  float* AS   = W + o; o += (size_t)NN * 4;
  float* AD   = W + o; o += (size_t)NN * 4;
  int*   IDX  = (int*)(W + o);                 // 4096 x 16 top-k indices

  // ---- 1. graph learning: h = tanh(window@gl_w + gl_b); sim = relu(h@h^T) ----
  launch_gemm(window, 256, gl_w, 256, 0, GLH, 256, NN, 256, 256,
              gl_b, nullptr, 1.f, ACT_TANH, stream);
  launch_gemm(GLH, 256, GLH, 256, 1, SIM, NN, NN, NN, 256,
              nullptr, nullptr, 1.f, ACT_RELU_DIAG, stream);
  topk16_kernel<<<NN, 256, 0, stream>>>(SIM, IDX);

  // ---- 2. encoder: 3 GAT layers, relu ----
  const float* x = window;
  float* bufs[2] = {X0, X1};
  int cur = 0;
  for (int i = 0; i < 3; i++) {
    launch_gemm(x, 256, enc_W + (size_t)i * 256 * 256, 256, 0, HLIN, 256,
                NN, 256, 256, nullptr, nullptr, 1.f, ACT_NONE, stream);
    gat_prep_kernel<<<NN, 256, 0, stream>>>(HLIN, enc_asrc + i * 256,
                                            enc_adst + i * 256, AS, AD, 4);
    gat_agg_kernel<<<NN, 256, 0, stream>>>(HLIN, IDX, AS, AD, enc_b + i * 256,
                                           bufs[cur], 4, 1, 256, 1);
    x = bufs[cur]; cur ^= 1;
  }
  float* h   = (float*)x;      // == X0 after 3 layers
  float* tmp = bufs[cur];      // == X1

  // ---- 3. transformer bottleneck (2 layers, 4 heads, dh=64) ----
  for (int l = 0; l < 2; l++) {
    launch_gemm(h, 256, tr_wqkv + (size_t)l * 768 * 256, 256, 1, HLIN, 768,
                NN, 768, 256, tr_bqkv + l * 768, nullptr, 1.f, ACT_NONE, stream);
    for (int hd = 0; hd < 4; hd++) {
      // scores = Q_h @ K_h^T   (Q/K sliced out of fused qkv with stride 768)
      launch_gemm(HLIN + hd * 64, 768, HLIN + 256 + hd * 64, 768, 1,
                  SIM, NN, NN, NN, 64, nullptr, nullptr, 1.f, ACT_NONE, stream);
      softmax_row_kernel<<<NN, 256, 0, stream>>>(SIM, NN, 0.125f);
      // O_h = softmax(scores) @ V_h  -> strided into merged O buffer
      launch_gemm(SIM, NN, HLIN + 512 + hd * 64, 768, 0,
                  OHB + hd * 64, 256, NN, 64, NN,
                  nullptr, nullptr, 1.f, ACT_NONE, stream);
    }
    // x = LN(x + O@wo^T + bo)
    launch_gemm(OHB, 256, tr_wo + (size_t)l * 256 * 256, 256, 1, tmp, 256,
                NN, 256, 256, tr_bo + l * 256, h, 1.f, ACT_NONE, stream);
    layernorm_kernel<<<NN, 256, 0, stream>>>(tmp, tr_ln1_g + l * 256, tr_ln1_b + l * 256);
    // x = LN(x + relu(x@w1+b1)@w2+b2)
    launch_gemm(tmp, 256, tr_w1 + (size_t)l * 256 * 512, 512, 0, HLIN, 512,
                NN, 512, 256, tr_b1 + l * 512, nullptr, 1.f, ACT_RELU, stream);
    launch_gemm(HLIN, 512, tr_w2 + (size_t)l * 512 * 256, 256, 0, h, 256,
                NN, 256, 512, tr_b2 + l * 256, tmp, 1.f, ACT_NONE, stream);
    layernorm_kernel<<<NN, 256, 0, stream>>>(h, tr_ln2_g + l * 256, tr_ln2_b + l * 256);
  }

  // ---- 4. skip: h = h + window@skip_w + skip_b ----
  launch_gemm(window, 256, skip_w, 256, 0, tmp, 256, NN, 256, 256,
              skip_b, h, 1.f, ACT_NONE, stream);

  // ---- 5. decoder: 2 GAT layers (relu) + final single-head GAT ----
  const float* dx = tmp;
  float* dnext = h;
  for (int i = 0; i < 2; i++) {
    launch_gemm(dx, 256, dec_W + (size_t)i * 256 * 256, 256, 0, HLIN, 256,
                NN, 256, 256, nullptr, nullptr, 1.f, ACT_NONE, stream);
    gat_prep_kernel<<<NN, 256, 0, stream>>>(HLIN, dec_asrc + i * 256,
                                            dec_adst + i * 256, AS, AD, 4);
    gat_agg_kernel<<<NN, 256, 0, stream>>>(HLIN, IDX, AS, AD, dec_b + i * 256,
                                           dnext, 4, 1, 256, 1);
    float* t = (float*)dx; dx = dnext; dnext = t;
  }
  launch_gemm(dx, 256, dec_lW, 256, 0, HLIN, 256, NN, 256, 256,
              nullptr, nullptr, 1.f, ACT_NONE, stream);
  gat_prep_kernel<<<NN, 256, 0, stream>>>(HLIN, dec_lasrc, dec_ladst, AS, AD, 1);
  // final GAT writes out.T directly: d_out[c*4096 + n]
  gat_agg_kernel<<<NN, 256, 0, stream>>>(HLIN, IDX, AS, AD, dec_lb,
                                         (float*)d_out, 1, 0, 1, NN);
}
